// _NonLocalBlock_75909251989704
// MI455X (gfx1250) — compile-verified, matrix operand order checked by reading, not run on hardware
//
#include <hip/hip_runtime.h>
#include <hip/hip_bf16.h>

// ---------------------------------------------------------------------------
// NonLocalBlock (B=4, C=256, CI=128, W=H=Z=32) for gfx1250 (MI455X, wave32).
// bf16 WMMA (v_wmma_f32_16x16x32_bf16) for all GEMMs, fp32 accumulate,
// fp32 softmax/batchnorm/residual. Async-to-LDS staging where layout permits.
// ---------------------------------------------------------------------------

typedef __attribute__((ext_vector_type(16))) __bf16 v16bf;
typedef __attribute__((ext_vector_type(8)))  float  v8f;

union Frag16 {
    v16bf v;
    uint4 q[2];
};

// fp32 -> bf16 round-to-nearest-even
static __device__ __forceinline__ unsigned short f2bf(float f) {
    unsigned u = __builtin_bit_cast(unsigned, f);
    u += 0x7FFFu + ((u >> 16) & 1u);
    return (unsigned short)(u >> 16);
}

#define NB   4
#define NC   256
#define NCI  128
#define NSP  32768          // 32*32*32
#define NWH  1024           // 32*32
#define NZ   32
#define NZP  16             // pooled z
#define NSPP 16384          // pooled spatial

// ---------------------------------------------------------------------------
// Kernel 1: transpose+convert x[b][c][n] f32 -> xT[b][n][c] bf16
// ---------------------------------------------------------------------------
__global__ __launch_bounds__(256) void k_cvt_xT(const float* __restrict__ x,
                                                unsigned short* __restrict__ xT) {
    __shared__ float tile[32][33];
    const int n0 = blockIdx.x * 32;
    const int c0 = blockIdx.y * 32;
    const int b  = blockIdx.z;
    const int tx = threadIdx.x, ty = threadIdx.y;   // block (32,8)
#pragma unroll
    for (int i = 0; i < 4; ++i) {
        int c = c0 + ty + i * 8;
        tile[ty + i * 8][tx] = x[((size_t)(b * NC + c) << 15) + n0 + tx];
    }
    __syncthreads();
#pragma unroll
    for (int i = 0; i < 4; ++i) {
        int n = n0 + ty + i * 8;
        xT[((size_t)(b * NSP + n) << 8) + c0 + tx] = f2bf(tile[tx][ty + i * 8]);
    }
}

// ---------------------------------------------------------------------------
// Kernel 2: convert weights. wqkv[384][256] = {g_w; theta_w; phi_w}, w2[256][128]
// ---------------------------------------------------------------------------
__global__ __launch_bounds__(256) void k_cvt_w(const float* __restrict__ gw,
                                               const float* __restrict__ tw,
                                               const float* __restrict__ pw,
                                               const float* __restrict__ ww,
                                               unsigned short* __restrict__ wqkv,
                                               unsigned short* __restrict__ w2) {
    int idx = blockIdx.x * 256 + threadIdx.x;
    if (idx < 384 * 256) {
        int row = idx >> 8, col = idx & 255;
        float v;
        if      (row < 128) v = gw[row * 256 + col];
        else if (row < 256) v = tw[(row - 128) * 256 + col];
        else                v = pw[(row - 256) * 256 + col];
        wqkv[idx] = f2bf(v);
    } else {
        int j = idx - 384 * 256;            // < 256*128
        w2[j] = f2bf(ww[j]);
    }
}

// ---------------------------------------------------------------------------
// Kernel 3: fused g/theta/phi conv as GEMM D(384 x 32768) = Wqkv * X per batch.
// One 16x32 tile per wave (A fragment reused for 2 wmma), K=256 in 8 steps.
// Epilogue: bias, z-pool for g/phi (shfl-pair max), bf16 stores.
// grid (128, 24, 4), block 256 (8 waves; wave w -> n-block blockIdx.x*8+w)
// ---------------------------------------------------------------------------
__global__ __launch_bounds__(256) void k_conv_qkv(
        const unsigned short* __restrict__ xT,    // [B][32768][256]
        const unsigned short* __restrict__ wqkv,  // [384][256]
        const float* __restrict__ gb, const float* __restrict__ tb,
        const float* __restrict__ pb,
        unsigned short* __restrict__ th,          // [B][128][32768]
        unsigned short* __restrict__ gx,          // [B][128][16384]
        unsigned short* __restrict__ ph)          // [B][128][16384]
{
    const int lane = threadIdx.x & 31;
    const int wave = threadIdx.x >> 5;
    const int row  = lane & 15;
    const int sel  = lane >> 4;

    const int b  = blockIdx.z;
    const int m0 = blockIdx.y * 16;
    const int n0 = (blockIdx.x * 8 + wave) * 32;

    const unsigned short* aBase  = wqkv + (size_t)(m0 + row) * NC;
    const unsigned short* b0Base = xT + (((size_t)b * NSP + n0 + row) << 8);
    const unsigned short* b1Base = xT + (((size_t)b * NSP + n0 + 16 + row) << 8);

    v8f acc0 = {}, acc1 = {};
#pragma unroll
    for (int k0 = 0; k0 < NC; k0 += 32) {
        Frag16 A, B0, B1;
        A.q[0]  = *(const uint4*)(aBase  + k0 + sel * 8);
        A.q[1]  = *(const uint4*)(aBase  + k0 + 16 + sel * 8);
        B0.q[0] = *(const uint4*)(b0Base + k0 + sel * 8);
        B0.q[1] = *(const uint4*)(b0Base + k0 + 16 + sel * 8);
        B1.q[0] = *(const uint4*)(b1Base + k0 + sel * 8);
        B1.q[1] = *(const uint4*)(b1Base + k0 + 16 + sel * 8);
        acc0 = __builtin_amdgcn_wmma_f32_16x16x32_bf16(
                false, A.v, false, B0.v, (short)0, acc0, false, false);
        acc1 = __builtin_amdgcn_wmma_f32_16x16x32_bf16(
                false, A.v, false, B1.v, (short)0, acc1, false, false);
    }

    const int kind = m0 >> 7;                 // 0=g, 1=theta, 2=phi
    const int cb   = m0 & 127;
    const float* bias = (kind == 0) ? gb : (kind == 1) ? tb : pb;

#pragma unroll
    for (int half = 0; half < 2; ++half) {
        const v8f& acc = half ? acc1 : acc0;
        const int  nb  = n0 + half * 16;
#pragma unroll
        for (int r = 0; r < 8; ++r) {
            int   c = cb + r + sel * 8;
            float v = acc[r] + bias[c];
            if (kind == 1) {
                th[((size_t)(b * NCI + c) << 15) + nb + row] = f2bf(v);
            } else {
                float o = __shfl_xor(v, 1, 32);   // pool z pairs (adjacent lanes)
                v = fmaxf(v, o);
                if ((lane & 1) == 0) {
                    unsigned short* dst = (kind == 0) ? gx : ph;
                    dst[((size_t)(b * NCI + c) << 14) + (nb >> 1) + (row >> 1)] = f2bf(v);
                }
            }
        }
    }
}

// ---------------------------------------------------------------------------
// Kernel 4: fused attention + second conv, one workgroup per (b, w, h) group.
// f = th^T ph (32x16,K=128) -> softmax -> y = gx attn^T (128x32) ->
// wy = W2 y + b (256x32) -> d_out.
// gx tile staged via GLOBAL_LOAD_ASYNC_TO_LDS_B32 (ASYNCcnt); th/ph tiles are
// transposed during staging so async copies don't apply there.
// grid (1024, 4), block 256 (8 waves)
// ---------------------------------------------------------------------------
__global__ __launch_bounds__(256) void k_attn(
        const unsigned short* __restrict__ th,   // [B][128][32768]
        const unsigned short* __restrict__ gx,   // [B][128][16384]
        const unsigned short* __restrict__ ph,   // [B][128][16384]
        const unsigned short* __restrict__ w2,   // [256][128]
        const float* __restrict__ wb,
        float* __restrict__ out)                 // [B][256][32768]
{
    __shared__ __align__(16) unsigned short th_t[32][136];   // [z][c]
    __shared__ __align__(16) unsigned short ph_t[16][136];   // [y][c]
    __shared__ __align__(16) unsigned short gx_s[128][40];   // [c][j], j>=16 zero
    __shared__ __align__(16) unsigned short at_t[32][40];    // [z][j], j>=16 zero
    __shared__ __align__(16) float          f_s[32][17];
    __shared__ __align__(16) unsigned short y_t[32][136];    // [z][c]

    const int tid  = threadIdx.x;
    const int lane = tid & 31;
    const int wave = tid >> 5;
    const int row  = lane & 15;
    const int sel  = lane >> 4;
    const int wh   = blockIdx.x;
    const int b    = blockIdx.y;

    // ---- async-stage gx tile (layout matches global: [c][j] linear) ----
    // 128 rows x 16 bf16 = 1024 dword transfers, 4 per thread.
    for (int i = tid; i < NCI * 8; i += 256) {
        int c = i >> 3, p = (i & 7) * 2;     // p = element pair start
        unsigned dstOff = (unsigned)(uintptr_t)&gx_s[c][p];
        unsigned long long srcAddr =
            (unsigned long long)(uintptr_t)(gx + ((size_t)(b * NCI + c) << 14) + wh * NZP + p);
        asm volatile("global_load_async_to_lds_b32 %0, %1, off"
                     :: "v"(dstOff), "v"(srcAddr) : "memory");
    }

    // ---- stage th/ph tiles into LDS transposed (K must be contiguous) ----
    for (int i = tid; i < NCI * NZ; i += 256) {
        int c = i >> 5, z = i & 31;
        th_t[z][c] = th[((size_t)(b * NCI + c) << 15) + wh * NZ + z];
    }
    for (int i = tid; i < NCI * NZP; i += 256) {
        int c = i >> 4, y = i & 15;
        ph_t[y][c]      = ph[((size_t)(b * NCI + c) << 14) + wh * NZP + y];
        gx_s[c][16 + y] = 0;                 // zero-pad K 16..31 of gx tile
    }
    asm volatile("s_wait_asynccnt 0x0" ::: "memory");
    __syncthreads();

    // ---- f = th^T * ph : M=32 (2 tiles, waves 0..1), N=16, K=128 ----
    if (wave < 2) {
        const int z0 = wave * 16;
        v8f fa = {};
#pragma unroll
        for (int k0 = 0; k0 < NCI; k0 += 32) {
            Frag16 A, Bm;
            A.q[0]  = *(const uint4*)&th_t[z0 + row][k0 + sel * 8];
            A.q[1]  = *(const uint4*)&th_t[z0 + row][k0 + 16 + sel * 8];
            Bm.q[0] = *(const uint4*)&ph_t[row][k0 + sel * 8];
            Bm.q[1] = *(const uint4*)&ph_t[row][k0 + 16 + sel * 8];
            fa = __builtin_amdgcn_wmma_f32_16x16x32_bf16(
                    false, A.v, false, Bm.v, (short)0, fa, false, false);
        }
#pragma unroll
        for (int r = 0; r < 8; ++r) f_s[z0 + r + 8 * sel][row] = fa[r];
    }
    __syncthreads();

    // ---- softmax over y (16) for each z row (fp32) ----
    if (tid < 32) {
        float m = -3.4e38f;
#pragma unroll
        for (int j = 0; j < 16; ++j) m = fmaxf(m, f_s[tid][j]);
        float s = 0.f, e[16];
#pragma unroll
        for (int j = 0; j < 16; ++j) { e[j] = __expf(f_s[tid][j] - m); s += e[j]; }
        float inv = 1.f / s;
#pragma unroll
        for (int j = 0; j < 16; ++j) {
            at_t[tid][j]      = f2bf(e[j] * inv);
            at_t[tid][16 + j] = 0;
        }
    }
    __syncthreads();

    // ---- y = gx * attn^T : M=128 (8 tiles), N=32 (2 tiles), K=16 (padded 32) ----
    {
        const int c0 = wave * 16;
#pragma unroll
        for (int i = 0; i < 2; ++i) {
            const int z0 = i * 16;
            Frag16 A, Bm;
            A.q[0]  = *(const uint4*)&gx_s[c0 + row][sel * 8];
            A.q[1]  = *(const uint4*)&gx_s[c0 + row][16 + sel * 8];
            Bm.q[0] = *(const uint4*)&at_t[z0 + row][sel * 8];
            Bm.q[1] = *(const uint4*)&at_t[z0 + row][16 + sel * 8];
            v8f ya = {};
            ya = __builtin_amdgcn_wmma_f32_16x16x32_bf16(
                    false, A.v, false, Bm.v, (short)0, ya, false, false);
#pragma unroll
            for (int r = 0; r < 8; ++r)
                y_t[z0 + row][c0 + r + 8 * sel] = f2bf(ya[r]);
        }
    }
    __syncthreads();

    // ---- wy = W2 * y + b : M=256 (16 tiles), N=32 (2 tiles), K=128 ----
#pragma unroll
    for (int i = 0; i < 4; ++i) {
        const int t  = wave * 4 + i;
        const int m0 = (t >> 1) * 16;
        const int z0 = (t & 1) * 16;
        const unsigned short* a2 = w2 + (size_t)(m0 + row) * NCI;
        v8f wa = {};
#pragma unroll
        for (int k0 = 0; k0 < NCI; k0 += 32) {
            Frag16 A, Bm;
            A.q[0]  = *(const uint4*)(a2 + k0 + sel * 8);
            A.q[1]  = *(const uint4*)(a2 + k0 + 16 + sel * 8);
            Bm.q[0] = *(const uint4*)&y_t[z0 + row][k0 + sel * 8];
            Bm.q[1] = *(const uint4*)&y_t[z0 + row][k0 + 16 + sel * 8];
            wa = __builtin_amdgcn_wmma_f32_16x16x32_bf16(
                    false, A.v, false, Bm.v, (short)0, wa, false, false);
        }
#pragma unroll
        for (int r = 0; r < 8; ++r) {
            int m = m0 + r + 8 * sel;
            out[((size_t)(b * NC + m) << 15) + wh * NZ + z0 + row] = wa[r] + wb[m];
        }
    }
}

// ---------------------------------------------------------------------------
// BatchNorm: stats + apply
// ---------------------------------------------------------------------------
__global__ __launch_bounds__(256) void k_zero_stats(float* __restrict__ stats) {
    int i = blockIdx.x * 256 + threadIdx.x;
    if (i < 512) stats[i] = 0.f;
}

__global__ __launch_bounds__(256) void k_stats(const float* __restrict__ out,
                                               float* __restrict__ stats) {
    __shared__ float rs[256], rq[256];
    const int bc = blockIdx.x;            // b*256 + c
    const int c  = bc & 255;
    const float* p = out + ((size_t)bc << 15);
    float s = 0.f, q = 0.f;
    for (int i = threadIdx.x; i < NSP; i += 256) {
        float v = p[i];
        s += v; q += v * v;
    }
    rs[threadIdx.x] = s; rq[threadIdx.x] = q;
    __syncthreads();
    for (int off = 128; off > 0; off >>= 1) {
        if (threadIdx.x < off) {
            rs[threadIdx.x] += rs[threadIdx.x + off];
            rq[threadIdx.x] += rq[threadIdx.x + off];
        }
        __syncthreads();
    }
    if (threadIdx.x == 0) {
        atomicAdd(&stats[c], rs[0]);
        atomicAdd(&stats[256 + c], rq[0]);
    }
}

__global__ __launch_bounds__(256) void k_bn_apply(const float* __restrict__ x,
                                                  const float* __restrict__ stats,
                                                  const float* __restrict__ gamma,
                                                  const float* __restrict__ beta,
                                                  float* __restrict__ out) {
    const float invN = 1.f / (float)(NB * NSP);
    const size_t total = (size_t)NB * NC * NSP;
    for (size_t idx = (size_t)blockIdx.x * 256 + threadIdx.x; idx < total;
         idx += (size_t)gridDim.x * 256) {
        int c = (int)((idx >> 15) & 255);
        float mean = stats[c] * invN;
        float var  = stats[256 + c] * invN - mean * mean;
        float v    = out[idx];
        out[idx] = (v - mean) * __frsqrt_rn(var + 1e-5f) * gamma[c] + beta[c] + x[idx];
    }
}

// ---------------------------------------------------------------------------
// Launch
// ---------------------------------------------------------------------------
extern "C" void kernel_launch(void* const* d_in, const int* in_sizes, int n_in,
                              void* d_out, int out_size, void* d_ws, size_t ws_size,
                              hipStream_t stream) {
    const float* x   = (const float*)d_in[0];
    const float* gw  = (const float*)d_in[1];
    const float* gb  = (const float*)d_in[2];
    const float* tw  = (const float*)d_in[3];
    const float* tb  = (const float*)d_in[4];
    const float* pw  = (const float*)d_in[5];
    const float* pb  = (const float*)d_in[6];
    const float* ww  = (const float*)d_in[7];
    const float* wb  = (const float*)d_in[8];
    const float* bng = (const float*)d_in[9];
    const float* bnb = (const float*)d_in[10];
    float* out = (float*)d_out;

    unsigned char* ws = (unsigned char*)d_ws;
    size_t off = 0;
    unsigned short* xT   = (unsigned short*)(ws + off); off += (size_t)NB * NSP * NC * 2;   // 64 MB
    unsigned short* wqkv = (unsigned short*)(ws + off); off += (size_t)384 * 256 * 2;
    unsigned short* w2   = (unsigned short*)(ws + off); off += (size_t)256 * 128 * 2;
    unsigned short* th   = (unsigned short*)(ws + off); off += (size_t)NB * NCI * NSP * 2;  // 32 MB
    unsigned short* gx   = (unsigned short*)(ws + off); off += (size_t)NB * NCI * NSPP * 2; // 16 MB
    unsigned short* ph   = (unsigned short*)(ws + off); off += (size_t)NB * NCI * NSPP * 2; // 16 MB
    float*          st   = (float*)(ws + off);

    k_cvt_xT<<<dim3(1024, 8, 4), dim3(32, 8), 0, stream>>>(x, xT);
    k_cvt_w<<<512, 256, 0, stream>>>(gw, tw, pw, ww, wqkv, w2);
    k_conv_qkv<<<dim3(128, 24, 4), 256, 0, stream>>>(xT, wqkv, gb, tb, pb, th, gx, ph);
    k_attn<<<dim3(1024, 4), 256, 0, stream>>>(th, gx, ph, w2, wb, out);
    k_zero_stats<<<2, 256, 0, stream>>>(st);
    k_stats<<<NB * NC, 256, 0, stream>>>(out, st);
    k_bn_apply<<<8192, 256, 0, stream>>>(x, st, bng, bnb, out);
}
